// MultiHeadAttention1d_68032281968842
// MI455X (gfx1250) — matrix operand order repaired and checked by reading, not compiled, not run on hardware
//
#include <hip/hip_runtime.h>
#include <hip/hip_bf16.h>

typedef __attribute__((ext_vector_type(16))) __bf16 v16bf;
typedef __attribute__((ext_vector_type(8)))  __bf16 v8bf;
typedef __attribute__((ext_vector_type(4)))  __bf16 v4bf;
typedef __attribute__((ext_vector_type(8)))  float  v8f;
typedef __attribute__((ext_vector_type(4)))  float  v4f;

#define B_   4
#define C_   256
#define N_   1024
#define H_   8
#define D_   256

// -----------------------------------------------------------------------------
// Kernel 1: 1x1-conv projection  out = W (OxC) * in (CxN) + bias   per batch.
// mode 0: store bf16 plain       [b][o][n]            (v)
// mode 1: store bf16 transposed  [b][h][n][d]         (q, k)
// mode 2: store f32 plain        [b][o][n]            (yp)
// Block: 256 thr (8 waves), tile 64(O) x 32(N), K loop 8 x 32, bf16 WMMA.
// -----------------------------------------------------------------------------
__global__ __launch_bounds__(256, 1)
void proj_kernel(const float* __restrict__ W, const float* __restrict__ bias,
                 const float* __restrict__ in, void* __restrict__ outp,
                 int O, int mode) {
    __shared__ __bf16 As[64][40];    // A tile row-major, 80B rows (16B aligned)
    __shared__ __bf16 Bst[32][48];   // B tile TRANSPOSED [n][k], 96B rows (32B aligned)
    const int tid = threadIdx.x;
    const int o0 = blockIdx.x * 64, n0 = blockIdx.y * 32, bb = blockIdx.z;
    const float* inB = in + (size_t)bb * C_ * N_;
    const int wid = tid >> 5, lane = tid & 31;
    const int wm = wid >> 1, wn = wid & 1;
    const int col = lane & 15, hi = lane >> 4;

    v8f cacc;
    #pragma unroll
    for (int r = 0; r < 8; ++r) cacc[r] = 0.0f;

    #pragma unroll 1
    for (int kk = 0; kk < C_; kk += 32) {
        // A tile: 2 x float4 per thread -> packed bf16 b64 LDS stores
        #pragma unroll
        for (int t = 0; t < 2; ++t) {
            int j = tid + t * 256;
            int r = j >> 3, c4 = (j & 7) << 2;
            v4f w4 = *(const v4f*)&W[(size_t)(o0 + r) * C_ + kk + c4];
            v4bf wb;
            #pragma unroll
            for (int i = 0; i < 4; ++i) wb[i] = (__bf16)w4[i];
            *(v4bf*)&As[r][c4] = wb;
        }
        // B tile: 1 x float4 per thread, transposed scatter into Bst
        {
            int r = tid >> 3, c4 = (tid & 7) << 2;
            v4f x4 = *(const v4f*)&inB[(size_t)(kk + r) * N_ + n0 + c4];
            #pragma unroll
            for (int i = 0; i < 4; ++i) Bst[c4 + i][r] = (__bf16)x4[i];
        }
        __syncthreads();

        v16bf a, b;
        {   // A-frag: two contiguous 16B runs per lane (ds_load_b128)
            v8bf lo = *(const v8bf*)&As[wm * 16 + col][16 * hi];
            v8bf hh = *(const v8bf*)&As[wm * 16 + col][16 * hi + 8];
            #pragma unroll
            for (int e = 0; e < 8; ++e) { a[e] = lo[e]; a[e + 8] = hh[e]; }
            // B-frag: one contiguous 32B run (K = e + 16*hi)
            b = *(const v16bf*)&Bst[wn * 16 + col][16 * hi];
        }
        cacc = __builtin_amdgcn_wmma_f32_16x16x32_bf16(false, a, false, b,
                                                       (short)0, cacc, false, false);
        __syncthreads();
    }

    const int o_lo = o0 + wm * 16 + 8 * hi;      // this lane's 8 consecutive rows
    const int n    = n0 + wn * 16 + col;
    if (mode == 1) {
        int hh2 = o_lo >> 8, dd = o_lo & 255;    // 8 consecutive dd -> one 16B store
        v8bf pk;
        #pragma unroll
        for (int r = 0; r < 8; ++r) pk[r] = (__bf16)(cacc[r] + bias[o_lo + r]);
        *(v8bf*)&((__bf16*)outp)[(((size_t)bb * (O >> 8) + hh2) * N_ + n) * D_ + dd] = pk;
    } else if (mode == 0) {
        #pragma unroll
        for (int r = 0; r < 8; ++r) {
            int o = o_lo + r;
            ((__bf16*)outp)[((size_t)bb * O + o) * N_ + n] = (__bf16)(cacc[r] + bias[o]);
        }
    } else {
        #pragma unroll
        for (int r = 0; r < 8; ++r) {
            int o = o_lo + r;
            ((float*)outp)[((size_t)bb * O + o) * N_ + n] = cacc[r];
        }
    }
}

// -----------------------------------------------------------------------------
// Kernel 2: flash attention + fused epilogue.
// Block: 128 thr (4 waves); each wave owns a 16-query tile of one (b,h).
// Online softmax over 32-key steps; output d-dim in 2 passes of 128.
// q/k from transposed ws [b][h][n][d]; v from plain ws [b][o][n].
// -----------------------------------------------------------------------------
__global__ __launch_bounds__(128, 1)
void attn_kernel(const __bf16* __restrict__ qT, const __bf16* __restrict__ kT,
                 const __bf16* __restrict__ vP, const float* __restrict__ yp,
                 const float* __restrict__ gamma, float* __restrict__ out) {
    __shared__ __bf16 Ps[4][16][40];             // per-wave P staging, 80B rows
    const int tid  = threadIdx.x;
    const int wid  = tid >> 5, lane = tid & 31;
    const int col  = lane & 15, hi = lane >> 4;
    const int bb   = blockIdx.z, h = blockIdx.y;
    const size_t bh = (size_t)bb * H_ + h;
    const int n0   = (blockIdx.x * 4 + wid) * 16;
    const float g  = gamma[h];
    const float inv1g = 1.0f / (1.0f + g);
    const float LOG2E = 1.4426950408889634f;

    // q^T fragments: two global b128 loads each (A-frag runs are contiguous)
    v16bf qa[8];
    const __bf16* qrow = qT + (bh * N_ + n0 + col) * D_;
    #pragma unroll
    for (int c = 0; c < 8; ++c) {
        v8bf lo = *(const v8bf*)(qrow + c * 32 + 16 * hi);
        v8bf hh = *(const v8bf*)(qrow + c * 32 + 16 * hi + 8);
        #pragma unroll
        for (int e = 0; e < 8; ++e) { qa[c][e] = lo[e]; qa[c][e + 8] = hh[e]; }
    }

    #pragma unroll 1
    for (int pass = 0; pass < 2; ++pass) {
        const int dbase = pass * 128;
        v8f acc[8];
        float m_r[8], l_r[8];
        #pragma unroll
        for (int t = 0; t < 8; ++t) {
            #pragma unroll
            for (int r = 0; r < 8; ++r) acc[t][r] = 0.0f;
        }
        #pragma unroll
        for (int r = 0; r < 8; ++r) { m_r[r] = -3.0e38f; l_r[r] = 0.0f; }

        #pragma unroll 1
        for (int m0 = 0; m0 < N_; m0 += 32) {
            // ---- scores S = q^T k for 32 keys (two 16x16 C tiles) ----
            v8f s0, s1;
            #pragma unroll
            for (int r = 0; r < 8; ++r) { s0[r] = 0.0f; s1[r] = 0.0f; }
            #pragma unroll
            for (int c = 0; c < 8; ++c) {
                const __bf16* kp0 = kT + (bh * N_ + m0 + col) * D_ + c * 32 + 16 * hi;
                v16bf k0 = *(const v16bf*)kp0;                    // 32B contiguous
                v16bf k1 = *(const v16bf*)(kp0 + (size_t)16 * D_);
                s0 = __builtin_amdgcn_wmma_f32_16x16x32_bf16(false, qa[c], false, k0,
                                                             (short)0, s0, false, false);
                s1 = __builtin_amdgcn_wmma_f32_16x16x32_bf16(false, qa[c], false, k1,
                                                             (short)0, s1, false, false);
            }
            // ---- online softmax (row reductions within 16-lane groups) ----
            float p0[8], p1[8], sc[8];
            #pragma unroll
            for (int r = 0; r < 8; ++r) {
                float mx = fmaxf(s0[r], s1[r]);
                #pragma unroll
                for (int msk = 1; msk < 16; msk <<= 1)
                    mx = fmaxf(mx, __shfl_xor(mx, msk, 32));
                float mn = fmaxf(m_r[r], mx);
                sc[r] = __builtin_exp2f((m_r[r] - mn) * LOG2E);
                p0[r] = __builtin_exp2f((s0[r] - mn) * LOG2E);
                p1[r] = __builtin_exp2f((s1[r] - mn) * LOG2E);
                float rs = p0[r] + p1[r];
                #pragma unroll
                for (int msk = 1; msk < 16; msk <<= 1)
                    rs += __shfl_xor(rs, msk, 32);
                l_r[r] = l_r[r] * sc[r] + rs;
                m_r[r] = mn;
            }
            #pragma unroll
            for (int t = 0; t < 8; ++t) {
                #pragma unroll
                for (int r = 0; r < 8; ++r) acc[t][r] *= sc[r];
            }
            // ---- re-fragment P: C layout -> A layout via wave-private LDS ----
            #pragma unroll
            for (int r = 0; r < 8; ++r) {
                Ps[wid][r + 8 * hi][col]      = (__bf16)p0[r];
                Ps[wid][r + 8 * hi][16 + col] = (__bf16)p1[r];
            }
            __syncthreads();
            v16bf pa;
            {
                v8bf lo = *(const v8bf*)&Ps[wid][col][16 * hi];
                v8bf hh = *(const v8bf*)&Ps[wid][col][16 * hi + 8];
                #pragma unroll
                for (int e = 0; e < 8; ++e) { pa[e] = lo[e]; pa[e + 8] = hh[e]; }
            }
            // ---- out^T += P * v^T : v plain layout makes B-frag contiguous ----
            #pragma unroll
            for (int t = 0; t < 8; ++t) {
                const __bf16* vchan =
                    vP + ((size_t)bb * (H_ * D_) + (size_t)h * D_ + dbase + t * 16 + col) * N_
                       + m0 + 16 * hi;
                v16bf vb = *(const v16bf*)vchan;                  // 32B contiguous
                acc[t] = __builtin_amdgcn_wmma_f32_16x16x32_bf16(false, pa, false, vb,
                                                                 (short)0, acc[t], false, false);
            }
            __syncthreads();
        }
        // ---- normalize + fused (g*attn + yp)/(1+g), vectorized rows ----
        float inv[8];
        #pragma unroll
        for (int r = 0; r < 8; ++r) inv[r] = 1.0f / l_r[r];
        #pragma unroll
        for (int t = 0; t < 8; ++t) {
            int d = dbase + t * 16 + col;
            const float* yrow = yp + ((size_t)bb * D_ + d) * N_ + n0 + 8 * hi;
            float* orow = out + (((size_t)bb * H_ + h) * (size_t)D_ + d) * N_ + n0 + 8 * hi;
            v4f y0 = *(const v4f*)yrow;
            v4f y1 = *(const v4f*)(yrow + 4);
            v4f r0, r1;
            #pragma unroll
            for (int r = 0; r < 4; ++r) {
                r0[r] = (g * acc[t][r]     * inv[r]     + y0[r]) * inv1g;
                r1[r] = (g * acc[t][r + 4] * inv[r + 4] + y1[r]) * inv1g;
            }
            *(v4f*)orow       = r0;
            *(v4f*)(orow + 4) = r1;
        }
    }
}

extern "C" void kernel_launch(void* const* d_in, const int* in_sizes, int n_in,
                              void* d_out, int out_size, void* d_ws, size_t ws_size,
                              hipStream_t stream) {
    (void)in_sizes; (void)n_in; (void)out_size; (void)ws_size;
    const float* x  = (const float*)d_in[0];
    const float* y  = (const float*)d_in[1];
    const float* Wq = (const float*)d_in[2];
    const float* bq = (const float*)d_in[3];
    const float* Wk = (const float*)d_in[4];
    const float* bk = (const float*)d_in[5];
    const float* Wv = (const float*)d_in[6];
    const float* bv = (const float*)d_in[7];
    const float* Wp = (const float*)d_in[8];
    const float* gm = (const float*)d_in[9];

    const size_t QS = (size_t)B_ * H_ * N_ * D_;   // 8M bf16 elements each
    __bf16* qT = (__bf16*)d_ws;
    __bf16* kT = qT + QS;
    __bf16* vP = kT + QS;
    float*  yp = (float*)(vP + QS);

    dim3 blk(256);
    proj_kernel<<<dim3((H_ * D_) / 64, N_ / 32, B_), blk, 0, stream>>>(Wq, bq, x, qT, H_ * D_, 1);
    proj_kernel<<<dim3((H_ * D_) / 64, N_ / 32, B_), blk, 0, stream>>>(Wk, bk, x, kT, H_ * D_, 1);
    proj_kernel<<<dim3((H_ * D_) / 64, N_ / 32, B_), blk, 0, stream>>>(Wv, bv, y, vP, H_ * D_, 0);
    proj_kernel<<<dim3(D_ / 64, N_ / 32, B_), blk, 0, stream>>>(Wp, nullptr, y, yp, D_, 2);
    attn_kernel<<<dim3(N_ / 64, H_, B_), dim3(128), 0, stream>>>(qT, kT, vP, yp, gm, (float*)d_out);
}
